// LSTMAutoEncoder_23922967839088
// MI455X (gfx1250) — matrix operand order repaired and checked by reading, not compiled, not run on hardware
//
#include <hip/hip_runtime.h>
#include <hip/hip_bf16.h>
#include <hip/hip_fp16.h>

typedef __attribute__((ext_vector_type(16))) _Float16 v16h;
typedef __attribute__((ext_vector_type(8)))  _Float16 v8h;
typedef __attribute__((ext_vector_type(8)))  float    v8f;

#define BB   128
#define LL   1024
#define CIN  32
#define CO   256
#define HH   512
#define GG   2048      // 4*H
#define KW   7
#define EPSV 1e-5f

// ---------------- conv stack ----------------

__global__ void conv1d_k(const float* __restrict__ x, const float* __restrict__ w,
                         float* __restrict__ y, int Cin, int Cout, int x_is_blc) {
    long idx = (long)blockIdx.x * blockDim.x + threadIdx.x;
    long total = (long)BB * Cout * LL;
    if (idx >= total) return;
    int l  = (int)(idx % LL);
    int co = (int)((idx / LL) % Cout);
    int b  = (int)(idx / ((long)LL * Cout));
    float acc = 0.f;
    for (int k = 0; k < KW; ++k) {
        int ll = l + k - KW / 2;
        if (ll < 0 || ll >= LL) continue;
        const float* wp = w + ((long)co * Cin) * KW + k;
        if (x_is_blc) {
            const float* xp = x + ((long)b * LL + ll) * Cin;
            for (int ci = 0; ci < Cin; ++ci) acc += xp[ci] * wp[(long)ci * KW];
        } else {
            const float* xp = x + ((long)b * Cin) * LL + ll;
            for (int ci = 0; ci < Cin; ++ci) acc += xp[(long)ci * LL] * wp[(long)ci * KW];
        }
    }
    y[idx] = acc;
}

__global__ void bnstats_k(const float* __restrict__ y, float* __restrict__ stats, int C) {
    int c = blockIdx.x;
    __shared__ float ssum[256];
    __shared__ float ssq[256];
    float s = 0.f, q = 0.f;
    const long N = (long)BB * LL;
    for (long i = threadIdx.x; i < N; i += 256) {
        long b = i / LL, l = i % LL;
        float v = y[(b * C + c) * LL + l];
        s += v; q += v * v;
    }
    ssum[threadIdx.x] = s; ssq[threadIdx.x] = q;
    __syncthreads();
    for (int off = 128; off > 0; off >>= 1) {
        if ((int)threadIdx.x < off) {
            ssum[threadIdx.x] += ssum[threadIdx.x + off];
            ssq[threadIdx.x]  += ssq[threadIdx.x + off];
        }
        __syncthreads();
    }
    if (threadIdx.x == 0) {
        float mean = ssum[0] / (float)N;
        float var  = ssq[0] / (float)N - mean * mean;
        stats[2 * c]     = mean;
        stats[2 * c + 1] = rsqrtf(var + EPSV);
    }
}

__global__ void bnrelu_k(float* __restrict__ y, const float* __restrict__ g,
                         const float* __restrict__ bb, const float* __restrict__ stats, int C) {
    long idx = (long)blockIdx.x * blockDim.x + threadIdx.x;
    long total = (long)BB * C * LL;
    if (idx >= total) return;
    int c = (int)((idx / LL) % C);
    float v = y[idx];
    v = g[c] * (v - stats[2 * c]) * stats[2 * c + 1] + bb[c];
    y[idx] = v > 0.f ? v : 0.f;
}

// conv3 output (NCL f32, staged in d_out) -> BN+ReLU -> f16 [B][L][C]
__global__ void bnrelu_tr_k(const float* __restrict__ y, const float* __restrict__ g,
                            const float* __restrict__ bb, const float* __restrict__ stats,
                            _Float16* __restrict__ xh) {
    long idx = (long)blockIdx.x * blockDim.x + threadIdx.x;
    long total = (long)BB * LL * CO;
    if (idx >= total) return;
    int c = (int)(idx % CO);
    long t = (idx / CO) % LL;
    long b = idx / ((long)CO * LL);
    float v = y[(b * CO + c) * LL + t];
    v = g[c] * (v - stats[2 * c]) * stats[2 * c + 1] + bb[c];
    xh[idx] = (_Float16)(v > 0.f ? v : 0.f);
}

// ---------------- LSTM weight prep ----------------
// Wc[k][n] (f16, [I+H][2048]) = k<I ? Wih[n][k] : Whh[n][k-I];  bias = bih+bhh
__global__ void prep_w_k(const float* __restrict__ Wih, const float* __restrict__ Whh,
                         const float* __restrict__ bih, const float* __restrict__ bhh,
                         _Float16* __restrict__ Wc, float* __restrict__ bias, int I) {
    long idx = (long)blockIdx.x * blockDim.x + threadIdx.x;
    long total = (long)(I + HH) * GG;
    if (idx >= total) return;
    int  n = (int)(idx % GG);
    long k = idx / GG;
    float v = (k < I) ? Wih[(long)n * I + k] : Whh[(long)n * HH + (k - I)];
    Wc[idx] = (_Float16)v;
    if (idx < GG) bias[idx] = bih[idx] + bhh[idx];
}

// ---------------- fused LSTM cell (WMMA, all-f16 operands, f32 accum/state) ----------------
// gates[128,2048] = [X|hprev] @ Wc (+bias), fused i/f/g/o epilogue.
// Double-buffered LDS, software-pipelined: tile t+1 global loads issue before tile t WMMAs;
// one barrier per k-tile.
// Grid: 32 blocks (16 hidden cols each), 256 thr = 8 waves, wave -> 16 batch rows.
template <int I>
__global__ void __launch_bounds__(256)
lstm_cell_k(const _Float16* __restrict__ xsrc, long xstride,
            const _Float16* __restrict__ hprev, _Float16* __restrict__ hnext,
            float* __restrict__ cst, const _Float16* __restrict__ Wc,
            const float* __restrict__ bias) {
    __shared__ _Float16 As[2][128][40];  // [buf][row][k]   stride 80B (16B-aligned vectors)
    __shared__ _Float16 Bs[2][64][40];   // [buf][col][k]   TRANSPOSED -> contiguous B frags

    constexpr int NT = (I + HH) / 32;    // k-tiles

    const int tid  = threadIdx.x;
    const int wv   = tid >> 5;
    const int lane = tid & 31;
    const int half = lane >> 4;
    const int lan  = lane & 15;
    const int n0   = blockIdx.x * 16;    // hidden-unit tile base

    // A staging: thread -> (row, 16 contiguous k)
    const int arow = tid >> 1;
    const int acol = (tid & 1) * 16;
    // B staging: thread -> (k row, 8 contiguous cols within one gate group)
    const int bk    = tid >> 3;
    const int bj    = (tid & 7) * 8;
    const int bgcol = (bj >> 4) * HH + n0 + (bj & 15);

    v8f acc[4];
#pragma unroll
    for (int g = 0; g < 4; ++g)
#pragma unroll
        for (int e = 0; e < 8; ++e) acc[g][e] = 0.f;

    v8h a0, a1, bw;  // in-flight staging registers

    auto load_regs = [&](int ti) {
        const int kb = ti * 32;
        const _Float16* src;
        long stride;
        int kk;
        if (kb < I) { src = xsrc;  stride = xstride; kk = kb; }
        else        { src = hprev; stride = HH;      kk = kb - I; }
        const v8h* p = (const v8h*)(src + (long)arow * stride + kk + acol);
        a0 = p[0];
        a1 = p[1];
        bw = *(const v8h*)(Wc + (long)(kb + bk) * GG + bgcol);
    };
    auto store_lds = [&](int buf) {
        *(v8h*)&As[buf][arow][acol]     = a0;
        *(v8h*)&As[buf][arow][acol + 8] = a1;
#pragma unroll
        for (int j = 0; j < 8; ++j) Bs[buf][bj + j][bk] = bw[j];
    };
    auto mma_tile = [&](int buf) {
        // A fragment: halves 0..7 = As[m][half*8 .. +7], halves 8..15 = As[m][16+half*8 .. +7]
        const int m = wv * 16 + lan;
        v8h al = *(const v8h*)&As[buf][m][half * 8];
        v8h ah = *(const v8h*)&As[buf][m][16 + half * 8];
        v16h af;
#pragma unroll
        for (int i = 0; i < 8; ++i) { af[i] = al[i]; af[8 + i] = ah[i]; }
#pragma unroll
        for (int g = 0; g < 4; ++g) {
            const int colj = g * 16 + lan;
            v8h bl = *(const v8h*)&Bs[buf][colj][half * 8];
            v8h bh = *(const v8h*)&Bs[buf][colj][16 + half * 8];
            v16h bf;
#pragma unroll
            for (int i = 0; i < 8; ++i) { bf[i] = bl[i]; bf[8 + i] = bh[i]; }
            acc[g] = __builtin_amdgcn_wmma_f32_16x16x32_f16(
                false, af, false, bf, (short)0, acc[g], false, false);
        }
    };

    // prologue: stage tile 0
    load_regs(0);
    store_lds(0);
    __syncthreads();

    for (int ti = 0; ti < NT; ++ti) {
        const int cur = ti & 1;
        if (ti + 1 < NT) load_regs(ti + 1);   // global loads overlap WMMAs below
        mma_tile(cur);
        if (ti + 1 < NT) store_lds(cur ^ 1);
        __syncthreads();  // staged buf ready for next iter; cur free for overwrite after
    }

    // epilogue: lane owns (m, n) for all four gates
    int n = n0 + lan;
    float bi = bias[n], bf_ = bias[HH + n], bg = bias[2 * HH + n], bo = bias[3 * HH + n];
#pragma unroll
    for (int r = 0; r < 8; ++r) {
        int  m    = wv * 16 + half * 8 + r;
        long cidx = (long)m * HH + n;
        float iv = acc[0][r] + bi;
        float fv = acc[1][r] + bf_;
        float gv = acc[2][r] + bg;
        float ov = acc[3][r] + bo;
        float si = 1.f / (1.f + __expf(-iv));
        float sf = 1.f / (1.f + __expf(-fv));
        float so = 1.f / (1.f + __expf(-ov));
        float cn = sf * cst[cidx] + si * tanhf(gv);
        cst[cidx]   = cn;
        hnext[cidx] = (_Float16)(so * tanhf(cn));
    }
}

// ---------------- decoder output projection (reads f16 h) ----------------
__global__ void proj_k(const _Float16* __restrict__ h0, const float* __restrict__ W,
                       const float* __restrict__ bb, float* __restrict__ out, int t) {
    int idx = blockIdx.x * blockDim.x + threadIdx.x;
    if (idx >= BB * CO) return;
    int n = idx % CO;
    int b = idx / CO;
    const _Float16* hp = h0 + (long)b * HH;
    const float*    wp = W + (long)n * HH;
    float acc = bb[n];
#pragma unroll 8
    for (int k = 0; k < HH; ++k) acc += (float)hp[k] * wp[k];
    out[((long)b * LL + t) * CO + n] = acc;
}

// ---------------- host ----------------

extern "C" void kernel_launch(void* const* d_in, const int* in_sizes, int n_in,
                              void* d_out, int out_size, void* d_ws, size_t ws_size,
                              hipStream_t stream) {
    (void)in_sizes; (void)n_in; (void)out_size; (void)ws_size;
    const float* x_in = (const float*)d_in[0];
    const float* w1 = (const float*)d_in[1];
    const float* g1 = (const float*)d_in[2];  const float* b1 = (const float*)d_in[3];
    const float* w2 = (const float*)d_in[4];
    const float* g2 = (const float*)d_in[5];  const float* b2 = (const float*)d_in[6];
    const float* w3 = (const float*)d_in[7];
    const float* g3 = (const float*)d_in[8];  const float* b3 = (const float*)d_in[9];
    const float* eWih0 = (const float*)d_in[10]; const float* eWhh0 = (const float*)d_in[11];
    const float* ebih0 = (const float*)d_in[12]; const float* ebhh0 = (const float*)d_in[13];
    const float* eWih1 = (const float*)d_in[14]; const float* eWhh1 = (const float*)d_in[15];
    const float* ebih1 = (const float*)d_in[16]; const float* ebhh1 = (const float*)d_in[17];
    const float* dWih0 = (const float*)d_in[18]; const float* dWhh0 = (const float*)d_in[19];
    const float* dbih0 = (const float*)d_in[20]; const float* dbhh0 = (const float*)d_in[21];
    const float* dWih1 = (const float*)d_in[22]; const float* dWhh1 = (const float*)d_in[23];
    const float* dbih1 = (const float*)d_in[24]; const float* dbhh1 = (const float*)d_in[25];
    const float* outW  = (const float*)d_in[26]; const float* outB  = (const float*)d_in[27];
    float* out = (float*)d_out;

    char* wp = (char*)d_ws;
    auto alloc = [&](size_t bytes) -> char* {
        char* p = wp;
        wp += (bytes + 255) & ~(size_t)255;
        return p;
    };
    _Float16* xh = (_Float16*)alloc((size_t)BB * LL * CO * 2);
    float* y1 = (float*)alloc((size_t)BB * 32 * LL * 4);
    float* y2 = (float*)alloc((size_t)BB * 64 * LL * 4);
    _Float16* Wce0 = (_Float16*)alloc((size_t)(CO + HH) * GG * 2);
    _Float16* Wce1 = (_Float16*)alloc((size_t)(HH + HH) * GG * 2);
    _Float16* Wcd0 = (_Float16*)alloc((size_t)(CO + HH) * GG * 2);
    _Float16* Wcd1 = (_Float16*)alloc((size_t)(HH + HH) * GG * 2);
    float* be0 = (float*)alloc(GG * 4);
    float* be1 = (float*)alloc(GG * 4);
    float* bd0 = (float*)alloc(GG * 4);
    float* bd1 = (float*)alloc(GG * 4);
    _Float16* h0a = (_Float16*)alloc((size_t)BB * HH * 2);
    _Float16* h0b = (_Float16*)alloc((size_t)BB * HH * 2);
    _Float16* h1a = (_Float16*)alloc((size_t)BB * HH * 2);
    _Float16* h1b = (_Float16*)alloc((size_t)BB * HH * 2);
    float* c0  = (float*)alloc((size_t)BB * HH * 4);
    float* c1  = (float*)alloc((size_t)BB * HH * 4);
    float* stats = (float*)alloc(2 * CO * 4);

    // --- conv stack (conv3 staged in d_out before final transpose) ---
    {
        long t1 = (long)BB * 32 * LL;
        conv1d_k<<<(int)((t1 + 255) / 256), 256, 0, stream>>>(x_in, w1, y1, CIN, 32, 1);
        bnstats_k<<<32, 256, 0, stream>>>(y1, stats, 32);
        bnrelu_k<<<(int)((t1 + 255) / 256), 256, 0, stream>>>(y1, g1, b1, stats, 32);

        long t2 = (long)BB * 64 * LL;
        conv1d_k<<<(int)((t2 + 255) / 256), 256, 0, stream>>>(y1, w2, y2, 32, 64, 0);
        bnstats_k<<<64, 256, 0, stream>>>(y2, stats, 64);
        bnrelu_k<<<(int)((t2 + 255) / 256), 256, 0, stream>>>(y2, g2, b2, stats, 64);

        long t3 = (long)BB * CO * LL;
        conv1d_k<<<(int)((t3 + 255) / 256), 256, 0, stream>>>(y2, w3, out, 64, CO, 0);
        bnstats_k<<<CO, 256, 0, stream>>>(out, stats, CO);
        bnrelu_tr_k<<<(int)((t3 + 255) / 256), 256, 0, stream>>>(out, g3, b3, stats, xh);
    }

    // --- LSTM weight prep (f32 -> f16 concatenated [Wih;Whh]^T, summed bias) ---
    {
        long te = (long)(CO + HH) * GG;  // K=768
        long th = (long)(HH + HH) * GG;  // K=1024
        prep_w_k<<<(int)((te + 255) / 256), 256, 0, stream>>>(eWih0, eWhh0, ebih0, ebhh0, Wce0, be0, CO);
        prep_w_k<<<(int)((th + 255) / 256), 256, 0, stream>>>(eWih1, eWhh1, ebih1, ebhh1, Wce1, be1, HH);
        prep_w_k<<<(int)((te + 255) / 256), 256, 0, stream>>>(dWih0, dWhh0, dbih0, dbhh0, Wcd0, bd0, CO);
        prep_w_k<<<(int)((th + 255) / 256), 256, 0, stream>>>(dWih1, dWhh1, dbih1, dbhh1, Wcd1, bd1, HH);
    }

    // --- zero initial states ---
    hipMemsetAsync(h0a, 0, (size_t)BB * HH * 2, stream);
    hipMemsetAsync(h1a, 0, (size_t)BB * HH * 2, stream);
    hipMemsetAsync(c0, 0, (size_t)BB * HH * 4, stream);
    hipMemsetAsync(c1, 0, (size_t)BB * HH * 4, stream);

    _Float16* h0buf[2] = {h0a, h0b};
    _Float16* h1buf[2] = {h1a, h1b};
    int p0 = 0, p1 = 0;
    const long xstride = (long)LL * CO;

    // --- encoder: t = 0 .. L-1 ---
    for (int t = 0; t < LL; ++t) {
        lstm_cell_k<CO><<<32, 256, 0, stream>>>(xh + (long)t * CO, xstride,
                                                h0buf[p0], h0buf[p0 ^ 1], c0, Wce0, be0);
        p0 ^= 1;
        lstm_cell_k<HH><<<32, 256, 0, stream>>>(h0buf[p0], (long)HH,
                                                h1buf[p1], h1buf[p1 ^ 1], c1, Wce1, be1);
        p1 ^= 1;
    }

    // --- decoder: t = L-1 .. 0, emit projection from h0 BEFORE stepping ---
    for (int t = LL - 1; t >= 0; --t) {
        proj_k<<<(BB * CO + 255) / 256, 256, 0, stream>>>(h0buf[p0], outW, outB, out, t);
        lstm_cell_k<CO><<<32, 256, 0, stream>>>(xh + (long)t * CO, xstride,
                                                h0buf[p0], h0buf[p0 ^ 1], c0, Wcd0, bd0);
        p0 ^= 1;
        lstm_cell_k<HH><<<32, 256, 0, stream>>>(h0buf[p0], (long)HH,
                                                h1buf[p1], h1buf[p1 ^ 1], c1, Wcd1, bd1);
        p1 ^= 1;
    }
}
